// SelfAttention_87479893885543
// MI455X (gfx1250) — matrix operand order repaired and checked by reading, not compiled, not run on hardware
//
#include <hip/hip_runtime.h>

typedef __attribute__((ext_vector_type(16))) _Float16 v16h;
typedef __attribute__((ext_vector_type(8)))  _Float16 v8h;
typedef __attribute__((ext_vector_type(8)))  float    v8f;
typedef __attribute__((ext_vector_type(4)))  unsigned int v4u;
typedef __attribute__((ext_vector_type(8)))  int      v8i;
typedef __attribute__((ext_vector_type(4)))  int      v4i;

constexpr int Bc = 8;
constexpr int Tc = 2048;
constexpr int Cc = 1024;
constexpr int Hc = 128;
constexpr int KSTEPS = Cc / 32;            // 32 K-steps in the projection GEMM

__device__ __forceinline__ v8f wmma_f16(v16h a, v16h b, v8f c) {
  // D = A(16x32 f16) * B(32x16 f16) + C(16x16 f32)
  return __builtin_amdgcn_wmma_f32_16x16x32_f16(
      /*neg_a=*/false, a, /*neg_b=*/false, b,
      /*c_mod=*/(short)0, c, /*reuse_a=*/false, /*reuse_b=*/false);
}

__device__ __forceinline__ v16h cat8(v8h lo, v8h hi) {
  return __builtin_shufflevector(lo, hi, 0, 1, 2, 3, 4, 5, 6, 7,
                                 8, 9, 10, 11, 12, 13, 14, 15);
}

// ---------------------------------------------------------------------------
// TDM: issue a 2D tensor tile load (f16 elements) into LDS.
// Builds the D# descriptor per CDNA5 ISA ch.8:
//   group0: count=1 | lds_addr | global_addr[56:0] | type=2
//   group1: data_size=1(2B) | tensor_dim0/1 | tile_dim0/1 | dim0_stride
// Tracked by TENSORcnt (S_WAIT_TENSORCNT); EXEC ignored; all operands uniform.
// ---------------------------------------------------------------------------
__device__ __forceinline__ void tdm_load_2d_f16(unsigned int lds_off,
                                                const void* gptr,
                                                unsigned int dim0,    // elems/row
                                                unsigned int dim1,    // rows
                                                unsigned int stride0, // elems
                                                unsigned int tile0,
                                                unsigned int tile1) {
  const unsigned long long ga = (unsigned long long)(size_t)gptr;
  v4u g0;
  g0[0] = 1u;                                          // count=1, user mode
  g0[1] = lds_off;                                     // LDS byte address
  g0[2] = (unsigned int)(ga & 0xFFFFFFFFu);            // global_addr[31:0]
  g0[3] = (unsigned int)((ga >> 32) & 0x1FFFFFFu)      // global_addr[56:32]
          | (2u << 30);                                // type = 2 ("image")
  v8i g1;
  g1[0] = (int)(1u << 16);                             // data_size=1 -> 2 bytes
  g1[1] = (int)((dim0 & 0xFFFFu) << 16);               // tensor_dim0[15:0]
  g1[2] = (int)((dim0 >> 16) | ((dim1 & 0xFFFFu) << 16)); // dim0 hi | dim1 lo
  g1[3] = (int)((dim1 >> 16) | (tile0 << 16));         // dim1 hi | tile_dim0
  g1[4] = (int)(tile1 & 0xFFFFu);                      // tile_dim1 (tile_dim2=0)
  g1[5] = (int)stride0;                                // tensor_dim0_stride lo32
  g1[6] = 0;                                           // stride0 hi | stride1 lo
  g1[7] = 0;
  v4i g2 = {0, 0, 0, 0};
  v4i g3 = {0, 0, 0, 0};
#if __clang_major__ >= 23
  v8i g4 = {0, 0, 0, 0, 0, 0, 0, 0};
  __builtin_amdgcn_tensor_load_to_lds(g0, g1, g2, g3, g4, 0);
#else
  __builtin_amdgcn_tensor_load_to_lds(g0, g1, g2, g3, 0);
#endif
}

__device__ __forceinline__ unsigned int lds_offset(const void* p) {
  // generic pointer to shared memory: low 32 bits are the LDS byte offset
  return (unsigned int)(size_t)p;
}

// ---------------------------------------------------------------------------
// Kernel 0: pack Wq/Wk/Wv (f32, [C,H]) into f16 WMMA B-fragment order:
//   Wp[mat][kstep][nt][lane][j]  with  value = W[kstep*32 + (lane>>4)*16 + j]
//                                             [nt*16 + (lane&15)]
// ---------------------------------------------------------------------------
__global__ __launch_bounds__(32)
void pack_w_kernel(const float* __restrict__ Wq, const float* __restrict__ Wk,
                   const float* __restrict__ Wv, _Float16* __restrict__ Wp) {
  const int kstep = blockIdx.x;
  const int nt    = blockIdx.y;
  const int mat   = blockIdx.z;
  const float* __restrict__ W = (mat == 0) ? Wq : (mat == 1) ? Wk : Wv;
  const int lane = threadIdx.x & 31;
  const int n = lane & 15, hi = lane >> 4;

  _Float16* dst = Wp + ((((size_t)mat * KSTEPS + kstep) * 8 + nt) * 32 + lane) * 16;
  const float* src = W + (size_t)(kstep * 32 + hi * 16) * Hc + nt * 16 + n;
#pragma unroll
  for (int j = 0; j < 16; ++j) dst[j] = (_Float16)src[(size_t)j * Hc];
}

// ---------------------------------------------------------------------------
// Kernel 1: fused QKV projection, one wave per (16-token tile, matrix).
//   q,k stored f16 [B*T, H]; v stored f16 transposed [B, H, T].
// ---------------------------------------------------------------------------
__global__ __launch_bounds__(32)
void qkv_proj_kernel(const float* __restrict__ x,
                     const _Float16* __restrict__ Wp,
                     _Float16* __restrict__ qh,
                     _Float16* __restrict__ kh,
                     _Float16* __restrict__ vTh) {
  const int mtile = blockIdx.x;
  const int mat   = blockIdx.y;            // 0=q, 1=k, 2=v
  const int lane = threadIdx.x & 31;
  const int n    = lane & 15;
  const int hi   = lane >> 4;

  const int arow = mtile * 16 + n;
  const _Float16* wbase = Wp + (size_t)mat * KSTEPS * 8 * 32 * 16 + (size_t)lane * 16;

  v8f acc[8] = {};

  for (int ks = 0; ks < KSTEPS; ++ks) {
    const float* xr = x + (size_t)arow * Cc + ks * 32 + hi * 16;
    const float4 xa = *(const float4*)(xr);
    const float4 xb = *(const float4*)(xr + 4);
    const float4 xc = *(const float4*)(xr + 8);
    const float4 xd = *(const float4*)(xr + 12);
    v16h a;
    a[0]  = (_Float16)xa.x; a[1]  = (_Float16)xa.y;
    a[2]  = (_Float16)xa.z; a[3]  = (_Float16)xa.w;
    a[4]  = (_Float16)xb.x; a[5]  = (_Float16)xb.y;
    a[6]  = (_Float16)xb.z; a[7]  = (_Float16)xb.w;
    a[8]  = (_Float16)xc.x; a[9]  = (_Float16)xc.y;
    a[10] = (_Float16)xc.z; a[11] = (_Float16)xc.w;
    a[12] = (_Float16)xd.x; a[13] = (_Float16)xd.y;
    a[14] = (_Float16)xd.z; a[15] = (_Float16)xd.w;

#pragma unroll
    for (int nt = 0; nt < 8; ++nt) {
      const v16h bf = *(const v16h*)(wbase + (((size_t)ks * 8 + nt) * 32) * 16);
      acc[nt] = wmma_f16(a, bf, acc[nt]);
    }
  }

  if (mat == 2) {
    // vT [B, H, T]: fixed h per lane, tokens consecutive -> contiguous v8h
    const int bidx = (mtile * 16) / Tc;
    const int t0   = (mtile * 16) % Tc + hi * 8;
#pragma unroll
    for (int nt = 0; nt < 8; ++nt) {
      v8h r;
#pragma unroll
      for (int v = 0; v < 8; ++v) r[v] = (_Float16)acc[nt][v];
      *(v8h*)(vTh + ((size_t)bidx * Hc + nt * 16 + n) * Tc + t0) = r;
    }
  } else {
    // q/k [B*T, H]: transpose through LDS for coalesced b128 stores
    __shared__ _Float16 tile[16 * Hc];
#pragma unroll
    for (int nt = 0; nt < 8; ++nt)
#pragma unroll
      for (int v = 0; v < 8; ++v)
        tile[(hi * 8 + v) * Hc + nt * 16 + n] = (_Float16)acc[nt][v];
    __syncthreads();
    _Float16* dst = ((mat == 0) ? qh : kh) + (size_t)(mtile * 16) * Hc;
    const int r  = lane >> 1;
    const int c0 = (lane & 1) * 64;
#pragma unroll
    for (int c = 0; c < 64; c += 8)
      *(v8h*)(dst + (size_t)r * Hc + c0 + c) =
          *(const v8h*)(tile + r * Hc + c0 + c);
  }
}

// ---------------------------------------------------------------------------
// Kernel 2: causal flash attention, one wave per (batch, 16-query tile).
// K/V blocks are staged into an LDS double buffer by the Tensor Data Mover
// (one tensor_load_to_lds per tile, double-buffered on TENSORcnt), WMMAs
// consume them via ds_load_b128. Scores never touch global memory.
//   S^T  = K_blk(32xH) @ Q_tile^T(Hx16)      -> lane = query (scalar m/l stats)
//   out^T += V^T(128x32) @ P^T(32x16)
// ---------------------------------------------------------------------------
__global__ __launch_bounds__(32)
void flash_attn_kernel(const _Float16* __restrict__ qh,
                       const _Float16* __restrict__ kh,
                       const _Float16* __restrict__ vTh,
                       float* __restrict__ out) {
  const int qt   = blockIdx.x;             // query tile, 0 .. T/16-1
  const int b    = blockIdx.y;             // batch
  const int lane = threadIdx.x & 31;
  const int n    = lane & 15;
  const int hi   = lane >> 4;

  const _Float16* qp = qh  + ((size_t)b * Tc + qt * 16) * Hc;
  const _Float16* kp = kh  + (size_t)b * Tc * Hc;
  const _Float16* vp = vTh + (size_t)b * Hc * Tc;

  // Double-buffered LDS staging: K block [32 keys][128 h], V block [128 h][32 k]
  __shared__ __align__(16) _Float16 ldsK[2][32 * Hc];
  __shared__ __align__(16) _Float16 ldsV[2][Hc * 32];

  const int nblk = (qt * 16 + 15) / 32 + 1; // key blocks up to causal diagonal

  // Prologue: TDM block 0 into buffer 0 (overlaps with the Q-fragment loads).
  tdm_load_2d_f16(lds_offset(&ldsK[0][0]), kp, Hc, Tc, Hc, Hc, 32);
  tdm_load_2d_f16(lds_offset(&ldsV[0][0]), vp, Tc, Hc, Tc, 32, Hc);

  // Preload Q^T B-fragments (reused for every key block):
  // lane column N = query n; K rows = c*32 + hi*8 .. +8, then c*32 + 16 + hi*8 .. +8
  v16h bq[4];
#pragma unroll
  for (int c = 0; c < 4; ++c) {
    const _Float16* p = qp + (size_t)n * Hc + c * 32 + hi * 8;
    bq[c] = cat8(*(const v8h*)(p), *(const v8h*)(p + 16));
  }

  v8f acc[8] = {};                          // out^T: 8 h-tiles, lane col = query
  float m_run = -__builtin_huge_valf();
  float l_run = 0.0f;
  const int qrow = qt * 16 + n;

  for (int kb = 0; kb < nblk; ++kb) {
    const int key0 = kb * 32;
    const int buf  = kb & 1;

    // Issue next block's TDM loads, then wait for the current block.
    // TDM ops complete in order: <=2 outstanding => block kb is resident.
    if (kb + 1 < nblk) {
      const int nbuf = buf ^ 1;
      tdm_load_2d_f16(lds_offset(&ldsK[nbuf][0]), kp + (size_t)(key0 + 32) * Hc,
                      Hc, Tc, Hc, Hc, 32);
      tdm_load_2d_f16(lds_offset(&ldsV[nbuf][0]), vp + key0 + 32,
                      Tc, Hc, Tc, 32, Hc);
      __builtin_amdgcn_s_wait_tensorcnt(2);
    } else {
      __builtin_amdgcn_s_wait_tensorcnt(0);
    }
    asm volatile("" ::: "memory");          // keep ds reads below the wait

    const _Float16* kbuf = &ldsK[buf][0];
    const _Float16* vbuf = &ldsV[buf][0];

    // ---- S^T = K_blk @ Q^T : two 16x16 C tiles (keys 0-15 / 16-31 of block)
    v8f st0 = {}, st1 = {};
#pragma unroll
    for (int c = 0; c < 4; ++c) {
      const _Float16* p0 = kbuf + (size_t)n * Hc + c * 32 + hi * 8;
      const _Float16* p1 = kbuf + (size_t)(16 + n) * Hc + c * 32 + hi * 8;
      const v16h a0 = cat8(*(const v8h*)(p0), *(const v8h*)(p0 + 16));
      const v16h a1 = cat8(*(const v8h*)(p1), *(const v8h*)(p1 + 16));
      st0 = wmma_f16(a0, bq[c], st0);
      st1 = wmma_f16(a1, bq[c], st1);
    }

    // C layout: lane col = query n; rows = keys key0+hi*8+v (st0), +16 (st1)
    const int kb0 = key0 + hi * 8;
    const int kb1 = key0 + 16 + hi * 8;

    float s0[8], s1[8];
    float pm = -__builtin_huge_valf();
#pragma unroll
    for (int v = 0; v < 8; ++v) {
      s0[v] = (kb0 + v <= qrow) ? st0[v] : -__builtin_huge_valf();
      s1[v] = (kb1 + v <= qrow) ? st1[v] : -__builtin_huge_valf();
      pm = fmaxf(pm, fmaxf(s0[v], s1[v]));
    }
    pm = fmaxf(pm, __shfl_xor(pm, 16, 32));

    const float m_new = fmaxf(m_run, pm);
    const float scale = __expf(m_run - m_new);   // exp(-inf)=0 on first block

    float p0v[8], p1v[8];
    float lsum = 0.0f;
#pragma unroll
    for (int v = 0; v < 8; ++v) {
      p0v[v] = __expf(s0[v] - m_new);
      p1v[v] = __expf(s1[v] - m_new);
      lsum += p0v[v] + p1v[v];
    }
    lsum += __shfl_xor(lsum, 16, 32);
    l_run = l_run * scale + lsum;
    m_run = m_new;

#pragma unroll
    for (int ht = 0; ht < 8; ++ht)
#pragma unroll
      for (int v = 0; v < 8; ++v) acc[ht][v] *= scale;

    // ---- Build P^T B-fragment (half-wave exchange via shfl_xor 16)
    v16h pb;
#pragma unroll
    for (int j = 0; j < 8; ++j) {
      const float send = (hi == 0) ? p1v[j] : p0v[j];
      const float recv = __shfl_xor(send, 16, 32);
      pb[j]     = (_Float16)((hi == 0) ? p0v[j] : recv);
      pb[j + 8] = (_Float16)((hi == 0) ? recv : p1v[j]);
    }

    // ---- out^T += V^T @ P^T  (A fragments from LDS [h][32] tile)
#pragma unroll
    for (int ht = 0; ht < 8; ++ht) {
      const _Float16* pv = vbuf + (size_t)(ht * 16 + n) * 32 + hi * 16;
      const v16h av = cat8(*(const v8h*)(pv), *(const v8h*)(pv + 8));
      acc[ht] = wmma_f16(av, pb, acc[ht]);
    }
  }

  // Epilogue: out[b, qrow, h] = acc / l
  const float inv_l = 1.0f / l_run;
  float* op = out + ((size_t)b * Tc + qrow) * Hc;
#pragma unroll
  for (int ht = 0; ht < 8; ++ht) {
    float4 r0, r1;
    r0.x = acc[ht][0] * inv_l; r0.y = acc[ht][1] * inv_l;
    r0.z = acc[ht][2] * inv_l; r0.w = acc[ht][3] * inv_l;
    r1.x = acc[ht][4] * inv_l; r1.y = acc[ht][5] * inv_l;
    r1.z = acc[ht][6] * inv_l; r1.w = acc[ht][7] * inv_l;
    *(float4*)(op + ht * 16 + hi * 8)     = r0;
    *(float4*)(op + ht * 16 + hi * 8 + 4) = r1;
  }
}

// ---------------------------------------------------------------------------
extern "C" void kernel_launch(void* const* d_in, const int* in_sizes, int n_in,
                              void* d_out, int out_size, void* d_ws, size_t ws_size,
                              hipStream_t stream) {
  const float* x  = (const float*)d_in[0];
  const float* Wq = (const float*)d_in[1];
  const float* Wk = (const float*)d_in[2];
  const float* Wv = (const float*)d_in[3];
  float* out = (float*)d_out;

  const size_t qkv_elems = (size_t)Bc * Tc * Hc;
  char* ws = (char*)d_ws;
  _Float16* qh  = (_Float16*)(ws);
  _Float16* kh  = (_Float16*)(ws + qkv_elems * sizeof(_Float16));
  _Float16* vTh = (_Float16*)(ws + 2 * qkv_elems * sizeof(_Float16));
  _Float16* Wp  = (_Float16*)(ws + 3 * qkv_elems * sizeof(_Float16));

  pack_w_kernel<<<dim3(KSTEPS, 8, 3), dim3(32), 0, stream>>>(Wq, Wk, Wv, Wp);

  qkv_proj_kernel<<<dim3(Bc * Tc / 16, 3), dim3(32), 0, stream>>>(
      x, Wp, qh, kh, vTh);

  flash_attn_kernel<<<dim3(Tc / 16, Bc), dim3(32), 0, stream>>>(
      qh, kh, vTh, out);

  (void)in_sizes; (void)n_in; (void)out_size; (void)ws_size;
}
